// LHC_50199577756275
// MI455X (gfx1250) — matrix operand order repaired and checked by reading, not compiled, not run on hardware
//
#include <hip/hip_runtime.h>
#include <hip/hip_bf16.h>
#include <math.h>

typedef __attribute__((ext_vector_type(16))) _Float16 v16h;
typedef __attribute__((ext_vector_type(8)))  _Float16 v8h;
typedef __attribute__((ext_vector_type(8)))  float    v8f;

#define B_    2
#define NF    48
#define NSTEP 47

__device__ __forceinline__ int refl(int i, int n) { return i < 0 ? -i : (i >= n ? 2*n - 2 - i : i); }

__device__ __forceinline__ v16h cat16(v8h lo, v8h hi) {
  return __builtin_shufflevector(lo, hi, 0,1,2,3,4,5,6,7,8,9,10,11,12,13,14,15);
}

__device__ __forceinline__ v8f wmma16(v16h a, v16h b, v8f c) {
  return __builtin_amdgcn_wmma_f32_16x16x32_f16(false, a, false, b, (short)0, c, false, false);
}

// A-matrix 16x32 f16 (ISA 7.12.2): lane row = lane&15; halves 0-7 hold K = 8*(lane>=16)+0..7,
// halves 8-15 hold K = 16 + 8*(lane>=16)+0..7  -> two contiguous 8-half runs.
// B-matrix 32x16 f16: lane col = lane&15; same K order as A -> two contiguous 8-half runs.

// ---------- weight convert + transpose to [tap][oc][ic] f16 ----------
__global__ void k_cvt_w(const float* __restrict__ src, _Float16* __restrict__ dst, int Cout, int Cin) {
  int t = blockIdx.x * blockDim.x + threadIdx.x;       // Cout*Cin*9
  if (t >= Cout * Cin * 9) return;
  int tap = t % 9, rem = t / 9, ic = rem % Cin, oc = rem / Cin;
  dst[(tap * Cout + oc) * Cin + ic] = (_Float16)src[t];
}

// ---------- encoder stage 1: x*2-1 -> conv3x3(3->32) reflect -> relu -> pool2 -> NHWC f16 ----------
__global__ void k_e1(const float* __restrict__ x, const float* __restrict__ w,
                     const float* __restrict__ bias, _Float16* __restrict__ h1) {
  int t = blockIdx.x * blockDim.x + threadIdx.x;       // B*64*64*32, c fastest
  if (t >= B_ * 64 * 64 * 32) return;
  int c = t & 31, ox = (t >> 5) & 63, oy = (t >> 11) & 63, b = t >> 17;
  float acc = 0.f;
  for (int dy = 0; dy < 2; ++dy)
    for (int dx = 0; dx < 2; ++dx) {
      int yy = oy * 2 + dy, xx = ox * 2 + dx;
      float s = bias[c];
      for (int ic = 0; ic < 3; ++ic)
        for (int ky = 0; ky < 3; ++ky)
          for (int kx = 0; kx < 3; ++kx) {
            int iy = refl(yy + ky - 1, 128), ix = refl(xx + kx - 1, 128);
            float xv = x[((b * 3 + ic) * 128 + iy) * 128 + ix] * 2.f - 1.f;
            s += w[((c * 3 + ic) * 3 + ky) * 3 + kx] * xv;
          }
      acc += fmaxf(s, 0.f);
    }
  h1[t] = (_Float16)(acc * 0.25f);                     // [b][oy][ox][c]
}

// ---------- 3x3 reflect conv, NHWC activations, implicit GEMM on WMMA ----------
// one wave = 16 oc x 16 pixels; weights [tap][oc][ic] staged to LDS as [tap][ocl][ic]
__global__ void k_conv3x3_wmma(const _Float16* __restrict__ in16, const _Float16* __restrict__ w16,
                               const float* __restrict__ bias,
                               _Float16* __restrict__ out16, float* __restrict__ out32,
                               int Cin, int Cout, int H, int W, int upsample, int do_relu) {
  __shared__ _Float16 wlds[9 * 16 * 64];
  const int tid = threadIdx.x;
  const int img = blockIdx.z;
  const int oc_base = blockIdx.y * 16;
  const int wave = tid >> 5, lane = tid & 31;
  const int nlane = lane & 15;
  const int off8 = (lane >> 4) << 3;                   // 0 or 8
  const int off16 = (lane >> 4) << 4;                  // 0 or 16

  // stage 9*16*Cin weights as b128 copies
  const int nv = (9 * 16 * Cin) >> 3;
  for (int i = tid; i < nv; i += blockDim.x) {
    int e = i << 3;
    int tap = e / (16 * Cin), rem = e % (16 * Cin);
    *(v8h*)(wlds + tap * 16 * Cin + rem) =
        *(const v8h*)(w16 + (tap * Cout + oc_base) * Cin + rem);
  }
  __syncthreads();

  const int p0 = (blockIdx.x * 4 + wave) * 16;
  const int y = p0 / W;
  const int x = (p0 % W) + nlane;
  const int inH = upsample ? (H >> 1) : H, inW = upsample ? (W >> 1) : W;

  v8f acc = {};
  for (int ky = 0; ky < 3; ++ky) {
    int yy = refl(y + ky - 1, H); if (upsample) yy >>= 1;
    for (int kx = 0; kx < 3; ++kx) {
      int xx = refl(x + kx - 1, W); if (upsample) xx >>= 1;
      int tap = ky * 3 + kx;
      const _Float16* bptr = in16 + ((size_t)(img * inH + yy) * inW + xx) * Cin;
      const _Float16* aptr = wlds + tap * 16 * Cin + nlane * Cin;
      for (int kc = 0; kc < Cin; kc += 32) {
        v8h alo = *(const v8h*)(aptr + kc + off8);
        v8h ahi = *(const v8h*)(aptr + kc + off8 + 16);
        v8h blo = *(const v8h*)(bptr + kc + off8);
        v8h bhi = *(const v8h*)(bptr + kc + off8 + 16);
        acc = wmma16(cat16(alo, ahi), cat16(blo, bhi), acc);
      }
    }
  }
  // C/D: row m = r + off8 (consecutive r -> consecutive oc) -> one b128 store in NHWC
  size_t obase = ((size_t)(img * H + y) * W + x) * Cout + oc_base + off8;
  if (out16) {
    v8h o;
#pragma unroll
    for (int r = 0; r < 8; ++r) {
      float v = acc[r] + bias[oc_base + off8 + r];
      if (do_relu) v = fmaxf(v, 0.f);
      o[r] = (_Float16)v;
    }
    *(v8h*)(out16 + obase) = o;
  }
  if (out32) {
    v8f o;
#pragma unroll
    for (int r = 0; r < 8; ++r) {
      float v = acc[r] + bias[oc_base + off8 + r];
      if (do_relu) v = fmaxf(v, 0.f);
      o[r] = v;
    }
    *(v8f*)(out32 + obase) = o;
  }
}

// ---------- pool e3 (NHWC 64x64 -> channel-major xp[b][c][n], f32 + f16) ----------
__global__ void k_pool(const float* __restrict__ fin, float* __restrict__ xp, _Float16* __restrict__ xp16) {
  int t = blockIdx.x * blockDim.x + threadIdx.x;       // B*32*1024
  if (t >= B_ * 32 * 1024) return;
  int n = t & 1023, c = (t >> 10) & 31, b = t >> 15;
  int oy = n >> 5, ox = n & 31;
  const float* base = fin + ((size_t)(b * 64 + 2 * oy) * 64 + 2 * ox) * 32 + c;
  float s = 0.25f * (base[0] + base[32] + base[64 * 32] + base[64 * 32 + 32]);
  xp[t] = s;
  xp16[t] = (_Float16)s;
}

__global__ void k_pos_init(float* __restrict__ pos) {
  int t = blockIdx.x * blockDim.x + threadIdx.x;       // B*1024
  if (t >= B_ * 1024) return;
  int n = t & 1023, b = t >> 10;
  pos[(b * 2 + 0) * 1024 + n] = -1.f + (2.f / 31.f) * (float)(n >> 5);
  pos[(b * 2 + 1) * 1024 + n] = -1.f + (2.f / 31.f) * (float)(n & 31);
}

// ---------- top-8 smallest inner products + gather-mean (fused, idx stays in regs) ----------
__global__ void k_topk_gather(const float* __restrict__ pos, const float* __restrict__ xp,
                              float* __restrict__ xg) {
  int t = blockIdx.x * blockDim.x + threadIdx.x;       // B*1024
  if (t >= B_ * 1024) return;
  int j = t & 1023, b = t >> 10;
  const float* px = pos + (b * 2 + 0) * 1024;
  const float* py = pos + (b * 2 + 1) * 1024;
  float pxj = px[j], pyj = py[j];
  float bv[8]; int bi[8];
#pragma unroll
  for (int k = 0; k < 8; ++k) { bv[k] = 3.4e38f; bi[k] = 0; }
  for (int i = 0; i < 1024; ++i) {
    float d = px[i] * pxj + py[i] * pyj;
    if (d < bv[7]) {                                   // strict < keeps earlier index on ties
      int k = 7;
      while (k > 0 && d < bv[k - 1]) { bv[k] = bv[k - 1]; bi[k] = bi[k - 1]; --k; }
      bv[k] = d; bi[k] = i;
    }
  }
  const float* xpb = xp + b * 32 * 1024;
  float* xgb = xg + b * 32 * 1024;
  for (int c = 0; c < 32; ++c) {
    float s = 0.f;
#pragma unroll
    for (int k = 0; k < 8; ++k) s += xpb[c * 1024 + bi[k]];
    xgb[c * 1024 + j] = s * 0.125f;
  }
}

// ---------- two tiny MLPs + pos update (f32 VALU) ----------
__global__ void k_mlp_update(const float* __restrict__ xg,
                             const float* __restrict__ r1w, const float* __restrict__ r1b,
                             const float* __restrict__ r2w, const float* __restrict__ r2b,
                             const float* __restrict__ v1w, const float* __restrict__ v1b,
                             const float* __restrict__ v2w, const float* __restrict__ v2b,
                             float* __restrict__ xp, _Float16* __restrict__ xp16,
                             float* __restrict__ pos) {
  int t = blockIdx.x * blockDim.x + threadIdx.x;       // B*1024
  if (t >= B_ * 1024) return;
  int n = t & 1023, b = t >> 10;
  const float* z = xg + b * 32 * 1024;
  float h1v[16];
  for (int o = 0; o < 16; ++o) {
    float s = r1b[o];
    for (int c = 0; c < 32; ++c) s += r1w[o * 32 + c] * z[c * 1024 + n];
    h1v[o] = fmaxf(s, 0.f);
  }
  float xn[32];
  for (int o = 0; o < 32; ++o) {
    float s = r2b[o];
    for (int c = 0; c < 16; ++c) s += r2w[o * 16 + c] * h1v[c];
    xn[o] = fmaxf(s, 0.f);
    xp[(b * 32 + o) * 1024 + n] = xn[o];
    xp16[(b * 32 + o) * 1024 + n] = (_Float16)xn[o];
  }
  float h2v[16];
  for (int o = 0; o < 16; ++o) {
    float s = v1b[o];
    for (int c = 0; c < 32; ++c) s += v1w[o * 32 + c] * xn[c];
    h2v[o] = fmaxf(s, 0.f);
  }
  for (int o = 0; o < 2; ++o) {
    float s = v2b[o];
    for (int c = 0; c < 16; ++c) s += v2w[o * 16 + c] * h2v[c];
    pos[(b * 2 + o) * 1024 + n] += tanhf(s);
  }
}

// ---------- Gaussian kernel, stored TRANSPOSED: kdT[b][p][n] = exp(-dist(n,p)*scale) ----------
__global__ void k_kd(const float* __restrict__ pos, _Float16* __restrict__ kdT) {
  int t = blockIdx.x * blockDim.x + threadIdx.x;       // B*1024*1024, n fastest
  if (t >= B_ * 1024 * 1024) return;
  int n = t & 1023, p = (t >> 10) & 1023, b = t >> 20;
  float rx = -1.f + (2.f / 31.f) * (float)(p >> 5);
  float ry = -1.f + (2.f / 31.f) * (float)(p & 31);
  float dx = pos[(b * 2 + 0) * 1024 + n] - rx;
  float dy = pos[(b * 2 + 1) * 1024 + n] - ry;
  kdT[t] = (_Float16)expf(-(dx * dx + dy * dy) * 0.02209708691f);
}

// ---------- sample: ys[img][p][c] = sum_n xp[c,n]*kd[n,p], WMMA K=1024, NHWC f16 out ----------
__global__ void k_sample(const _Float16* __restrict__ xp16, const _Float16* __restrict__ kdT,
                         _Float16* __restrict__ ys16, int step) {
  const int lane = threadIdx.x & 31, wave = threadIdx.x >> 5;
  const int b = blockIdx.z;
  const int oc_base = blockIdx.y * 16;
  const int p0 = (blockIdx.x * 4 + wave) * 16;
  const int nlane = lane & 15;
  const int off8 = (lane >> 4) << 3, off16 = (lane >> 4) << 4;
  const _Float16* arow = xp16 + (size_t)(b * 32 + oc_base + nlane) * 1024;       // A: row=c, K=n
  const _Float16* brow = kdT + (size_t)b * 1024 * 1024 + (size_t)(p0 + nlane) * 1024; // B: col=p, K=n
  v8f acc = {};
  for (int k0 = 0; k0 < 1024; k0 += 32) {
    v8h alo = *(const v8h*)(arow + k0 + off8);
    v8h ahi = *(const v8h*)(arow + k0 + off8 + 16);
    v8h blo = *(const v8h*)(brow + k0 + off8);
    v8h bhi = *(const v8h*)(brow + k0 + off8 + 16);
    acc = wmma16(cat16(alo, ahi), cat16(blo, bhi), acc);
  }
  v8h o;
#pragma unroll
  for (int r = 0; r < 8; ++r) o[r] = (_Float16)acc[r];
  size_t img = (size_t)b * NSTEP + step;
  *(v8h*)(ys16 + (img * 1024 + p0 + nlane) * 32 + oc_base + off8) = o;
}

// ---------- d3: up2 + conv3x3(32->3, NHWC in) + tanh + rescale, frames 1..47 ----------
__global__ void k_d3(const _Float16* __restrict__ hd2, const float* __restrict__ w,
                     const float* __restrict__ bias, float* __restrict__ out) {
  int t = blockIdx.x * blockDim.x + threadIdx.x;       // 94*3*128*128
  if (t >= 94 * 3 * 128 * 128) return;
  int xx = t & 127, yy = (t >> 7) & 127, oc = (t >> 14) % 3, img = t / (3 * 128 * 128);
  float s = bias[oc];
  for (int ky = 0; ky < 3; ++ky) {
    int iy = refl(yy + ky - 1, 128) >> 1;
    for (int kx = 0; kx < 3; ++kx) {
      int ix = refl(xx + kx - 1, 128) >> 1;
      const v8h* ipv = (const v8h*)(hd2 + ((size_t)img * 4096 + iy * 64 + ix) * 32);
      const float* wp = w + oc * 32 * 9 + ky * 3 + kx;
#pragma unroll
      for (int ch = 0; ch < 4; ++ch) {
        v8h v = ipv[ch];
#pragma unroll
        for (int e = 0; e < 8; ++e) s += wp[(ch * 8 + e) * 9] * (float)v[e];
      }
    }
  }
  int b = img / NSTEP, f = img % NSTEP + 1;
  out[(((b * NF + f) * 3 + oc) * 128 + yy) * 128 + xx] = (tanhf(s) + 1.f) * 0.5f;
}

// ---------- frame 0 = original x ----------
__global__ void k_first(const float* __restrict__ x, float* __restrict__ out) {
  int t = blockIdx.x * blockDim.x + threadIdx.x;       // B*3*16384
  if (t >= B_ * 3 * 16384) return;
  int p = t % (3 * 16384), b = t / (3 * 16384);
  out[(size_t)b * NF * 3 * 16384 + p] = x[t];
}

extern "C" void kernel_launch(void* const* d_in, const int* in_sizes, int n_in,
                              void* d_out, int out_size, void* d_ws, size_t ws_size,
                              hipStream_t stream) {
  (void)in_sizes; (void)n_in; (void)out_size; (void)ws_size;
  const float* x   = (const float*)d_in[0];
  const float* e1w = (const float*)d_in[1];  const float* e1b = (const float*)d_in[2];
  const float* e2w = (const float*)d_in[3];  const float* e2b = (const float*)d_in[4];
  const float* e3w = (const float*)d_in[5];  const float* e3b = (const float*)d_in[6];
  const float* r1w = (const float*)d_in[7];  const float* r1b = (const float*)d_in[8];
  const float* r2w = (const float*)d_in[9];  const float* r2b = (const float*)d_in[10];
  const float* v1w = (const float*)d_in[11]; const float* v1b = (const float*)d_in[12];
  const float* v2w = (const float*)d_in[13]; const float* v2b = (const float*)d_in[14];
  const float* d1w = (const float*)d_in[15]; const float* d1b = (const float*)d_in[16];
  const float* d2w = (const float*)d_in[17]; const float* d2b = (const float*)d_in[18];
  const float* d3w = (const float*)d_in[19]; const float* d3b = (const float*)d_in[20];
  float* out = (float*)d_out;

  char* wsp = (char*)d_ws;
  auto carve = [&](size_t bytes) -> void* {
    void* p = (void*)wsp; wsp += (bytes + 255) & ~(size_t)255; return p;
  };
  _Float16* w16_e2 = (_Float16*)carve((size_t)64 * 32 * 9 * 2);
  _Float16* w16_e3 = (_Float16*)carve((size_t)32 * 64 * 9 * 2);
  _Float16* w16_d1 = (_Float16*)carve((size_t)64 * 32 * 9 * 2);
  _Float16* w16_d2 = (_Float16*)carve((size_t)32 * 64 * 9 * 2);
  _Float16* h1_16  = (_Float16*)carve((size_t)B_ * 64 * 64 * 32 * 2);   // NHWC
  _Float16* h2_16  = (_Float16*)carve((size_t)B_ * 64 * 64 * 64 * 2);   // NHWC
  float*    e3full = (float*)   carve((size_t)B_ * 64 * 64 * 32 * 4);   // NHWC f32
  float*    xp     = (float*)   carve((size_t)B_ * 32 * 1024 * 4);
  _Float16* xp16   = (_Float16*)carve((size_t)B_ * 32 * 1024 * 2);
  float*    xg     = (float*)   carve((size_t)B_ * 32 * 1024 * 4);
  float*    pos    = (float*)   carve((size_t)B_ * 2 * 1024 * 4);
  _Float16* kdT16  = (_Float16*)carve((size_t)B_ * 1024 * 1024 * 2);    // [b][p][n]
  _Float16* ys16   = (_Float16*)carve((size_t)B_ * NSTEP * 1024 * 32 * 2); // NHWC
  _Float16* hd1_16 = (_Float16*)carve((size_t)94 * 64 * 64 * 64 * 2);   // NHWC
  _Float16* hd2_16 = (_Float16*)carve((size_t)94 * 64 * 64 * 32 * 2);   // NHWC

  // weights -> f16, transposed to [tap][oc][ic]
  k_cvt_w<<<(64 * 32 * 9 + 255) / 256, 256, 0, stream>>>(e2w, w16_e2, 64, 32);
  k_cvt_w<<<(32 * 64 * 9 + 255) / 256, 256, 0, stream>>>(e3w, w16_e3, 32, 64);
  k_cvt_w<<<(64 * 32 * 9 + 255) / 256, 256, 0, stream>>>(d1w, w16_d1, 64, 32);
  k_cvt_w<<<(32 * 64 * 9 + 255) / 256, 256, 0, stream>>>(d2w, w16_d2, 32, 64);

  // encoder
  k_e1<<<(B_ * 64 * 64 * 32 + 255) / 256, 256, 0, stream>>>(x, e1w, e1b, h1_16);
  k_conv3x3_wmma<<<dim3(64, 4, B_), 128, 0, stream>>>(h1_16, w16_e2, e2b, h2_16, nullptr, 32, 64, 64, 64, 0, 1);
  k_conv3x3_wmma<<<dim3(64, 2, B_), 128, 0, stream>>>(h2_16, w16_e3, e3b, nullptr, e3full, 64, 32, 64, 64, 0, 1);
  k_pool<<<(B_ * 32 * 1024 + 255) / 256, 256, 0, stream>>>(e3full, xp, xp16);
  k_pos_init<<<(B_ * 1024 + 255) / 256, 256, 0, stream>>>(pos);

  // 47-step point-cloud scan
  for (int s = 0; s < NSTEP; ++s) {
    k_topk_gather<<<(B_ * 1024 + 255) / 256, 256, 0, stream>>>(pos, xp, xg);
    k_mlp_update<<<(B_ * 1024 + 255) / 256, 256, 0, stream>>>(xg, r1w, r1b, r2w, r2b,
                                                              v1w, v1b, v2w, v2b, xp, xp16, pos);
    k_kd<<<(B_ * 1024 * 1024 + 255) / 256, 256, 0, stream>>>(pos, kdT16);
    k_sample<<<dim3(16, 2, B_), 128, 0, stream>>>(xp16, kdT16, ys16, s);
  }

  // decoder (up2 folded into conv addressing; ys16 is NHWC 32x32x32 per image)
  k_conv3x3_wmma<<<dim3(64, 4, 94), 128, 0, stream>>>(ys16, w16_d1, d1b, hd1_16, nullptr, 32, 64, 64, 64, 1, 1);
  k_conv3x3_wmma<<<dim3(64, 2, 94), 128, 0, stream>>>(hd1_16, w16_d2, d2b, hd2_16, nullptr, 64, 32, 64, 64, 0, 1);
  k_d3<<<(94 * 3 * 128 * 128 + 255) / 256, 256, 0, stream>>>(hd2_16, d3w, d3b, out);
  k_first<<<(B_ * 3 * 16384 + 255) / 256, 256, 0, stream>>>(x, out);
}
